// MultiHeadAttention_10900626997682
// MI455X (gfx1250) — matrix operand order repaired and checked by reading, not compiled, not run on hardware
//
#include <hip/hip_runtime.h>
#include <hip/hip_bf16.h>

// Problem constants (reference: B=2, S=2048, D=1024, H=16, HD=64)
#define BB 2
#define SS 2048
#define DD 1024
#define HH 16
#define HDIM 64
#define MROWS (BB * SS)   // 4096
#define NQKV (3 * DD)     // 3072

typedef __bf16 bf16;
typedef __attribute__((ext_vector_type(16))) __bf16 v16bf;
typedef __attribute__((ext_vector_type(8)))  __bf16 v8bf;
typedef __attribute__((ext_vector_type(8)))  float  v8f;

union Frag {
  v16bf v;
  v8bf  h[2];
};

__device__ __forceinline__ v8f wmma_bf16(v16bf a, v16bf b, v8f c) {
  // D = A(16x32 bf16) * B(32x16 bf16) + C(16x16 f32)
  return __builtin_amdgcn_wmma_f32_16x16x32_bf16(
      /*neg_a=*/false, a, /*neg_b=*/false, b,
      /*c_mod=*/(short)0, c, /*reuse_a=*/false, /*reuse_b=*/false);
}

// ---------------------------------------------------------------------------
// Elementwise fp32 -> bf16 convert
// ---------------------------------------------------------------------------
__global__ void __launch_bounds__(256)
convert_bf16_kernel(const float* __restrict__ in, bf16* __restrict__ out, int n) {
  int i = blockIdx.x * blockDim.x + threadIdx.x;
  if (i < n) out[i] = (bf16)in[i];
}

// ---------------------------------------------------------------------------
// Transpose fp32 (K x N) -> bf16 (N x K), writes coalesced
// ---------------------------------------------------------------------------
__global__ void __launch_bounds__(256)
transpose_bf16_kernel(const float* __restrict__ W, bf16* __restrict__ Wt, int K, int N) {
  size_t i = (size_t)blockIdx.x * blockDim.x + threadIdx.x;
  if (i < (size_t)K * N) {
    int n = (int)(i / (size_t)K);
    int k = (int)(i % (size_t)K);
    Wt[i] = (bf16)W[(size_t)k * N + n];
  }
}

// ---------------------------------------------------------------------------
// 64x64-per-wave bf16 MMA core. A is (M x K) row-major (lda = K),
// Bt is (N x K) row-major (pre-transposed weights), acc is 4x4 WMMA tiles.
// ---------------------------------------------------------------------------
__device__ __forceinline__ void mma_64x64(const bf16* __restrict__ A,
                                          const bf16* __restrict__ Bt,
                                          int m0, int n0, int K, v8f acc[4][4]) {
  const int lane = threadIdx.x & 31;
  const int l16  = lane & 15;
  const int g    = lane >> 4;  // half-wave group
  for (int kk = 0; kk < K; kk += 32) {
    Frag a[4], b[4];
#pragma unroll
    for (int i = 0; i < 4; ++i) {
      // A fragment: row m0+16i+l16, K elements e<8 -> kk+8g.., e>=8 -> kk+16+8g..
      const bf16* ap = A + (size_t)(m0 + 16 * i + l16) * K + kk + 8 * g;
      a[i].h[0] = *(const v8bf*)ap;
      a[i].h[1] = *(const v8bf*)(ap + 16);
    }
#pragma unroll
    for (int j = 0; j < 4; ++j) {
      // B fragment: column n0+16j+l16, K elements e<8 -> kk+8g.., e>=8 -> kk+16+8g..
      const bf16* bp = Bt + (size_t)(n0 + 16 * j + l16) * K + kk + 8 * g;
      b[j].h[0] = *(const v8bf*)bp;
      b[j].h[1] = *(const v8bf*)(bp + 16);
    }
#pragma unroll
    for (int i = 0; i < 4; ++i)
#pragma unroll
      for (int j = 0; j < 4; ++j)
        acc[i][j] = wmma_bf16(a[i].v, b[j].v, acc[i][j]);
  }
}

// ---------------------------------------------------------------------------
// Fused QKV projection: Xbf(4096x1024) @ WtQKV^T + b_qkv.
// Epilogue scatters: Q (scaled 1/sqrt(HD)) -> (B,H,S,HD), K -> (B,H,S,HD),
// V -> transposed (B,H,HD,S) so attention V B-fragments are contiguous.
// ---------------------------------------------------------------------------
__global__ void __launch_bounds__(128)
qkv_gemm_kernel(const bf16* __restrict__ Xb, const bf16* __restrict__ Wt,
                const float* __restrict__ bqkv,
                bf16* __restrict__ Qb, bf16* __restrict__ Kb, bf16* __restrict__ Vt) {
  const int wave = blockIdx.x * (blockDim.x >> 5) + (threadIdx.x >> 5);
  const int NT = NQKV / 64;  // 48
  const int MT = MROWS / 64; // 64
  if (wave >= MT * NT) return;
  const int tj = wave % NT;
  const int ti = wave / NT;
  const int m0 = ti * 64, n0 = tj * 64;
  const int lane = threadIdx.x & 31;
  const int l16  = lane & 15;
  const int g    = lane >> 4;

  v8f acc[4][4];
#pragma unroll
  for (int i = 0; i < 4; ++i)
#pragma unroll
    for (int j = 0; j < 4; ++j) { v8f z = {}; acc[i][j] = z; }

  mma_64x64(Xb, Wt, m0, n0, DD, acc);

#pragma unroll
  for (int j = 0; j < 4; ++j) {
    const int n    = n0 + 16 * j + l16;     // column in [0, 3072)
    const float bn = bqkv[n];
    const int part = n >> 10;               // 0=Q, 1=K, 2=V
    const int nn   = n & (DD - 1);
    const int hh   = nn >> 6;               // head
    const int dd   = nn & (HDIM - 1);       // dim within head
#pragma unroll
    for (int i = 0; i < 4; ++i) {
#pragma unroll
      for (int r = 0; r < 8; ++r) {
        const int m  = m0 + 16 * i + r + 8 * g; // row in [0, 4096)
        const int bb = m >> 11;                 // batch (S = 2048)
        const int s  = m & (SS - 1);
        const float val = acc[i][j][r] + bn;
        if (part == 0) {
          Qb[(((size_t)bb * HH + hh) * SS + s) * HDIM + dd] = (bf16)(val * 0.125f);
        } else if (part == 1) {
          Kb[(((size_t)bb * HH + hh) * SS + s) * HDIM + dd] = (bf16)val;
        } else {
          Vt[(((size_t)bb * HH + hh) * HDIM + dd) * SS + s] = (bf16)val;
        }
      }
    }
  }
}

// ---------------------------------------------------------------------------
// Causal flash attention. One wave handles one 16-query tile of one (b,h).
// Scores via Q@K^T WMMA (32 keys/chunk), online softmax, P re-laid-out via
// per-wave LDS, context via P@V WMMA.  Output -> bf16 context (B,S,H,HD).
// ---------------------------------------------------------------------------
__global__ void __launch_bounds__(128)
attn_kernel(const bf16* __restrict__ Qb, const bf16* __restrict__ Kb,
            const bf16* __restrict__ Vt, bf16* __restrict__ Ctx) {
  __shared__ __bf16 sP[4][16][32];  // per-wave P tile (16 queries x 32 keys)

  const int w    = threadIdx.x >> 5;
  const int lane = threadIdx.x & 31;
  const int l16  = lane & 15;
  const int g    = lane >> 4;
  const int wave = blockIdx.x * 4 + w;

  const int qt = wave & (SS / 16 - 1);  // 128 query tiles per (b,h)
  const int bh = wave >> 7;
  const int h  = bh & (HH - 1);
  const int b  = bh >> 4;
  const int q0 = qt * 16;

  const bf16* Qp = Qb + ((size_t)(b * HH + h) * SS) * HDIM;
  const bf16* Kp = Kb + ((size_t)(b * HH + h) * SS) * HDIM;
  const bf16* Vp = Vt + ((size_t)(b * HH + h) * HDIM) * SS;

  // Q A-fragments (16 x 64, two K-steps of 32), loaded once
  Frag qa[2];
#pragma unroll
  for (int kd = 0; kd < 2; ++kd) {
    const bf16* qp = Qp + (size_t)(q0 + l16) * HDIM + 32 * kd + 8 * g;
    qa[kd].h[0] = *(const v8bf*)qp;
    qa[kd].h[1] = *(const v8bf*)(qp + 16);
  }

  float mrow[8], lrow[8];
  v8f ctx[4];
#pragma unroll
  for (int r = 0; r < 8; ++r) { mrow[r] = -1e30f; lrow[r] = 0.0f; }
#pragma unroll
  for (int t = 0; t < 4; ++t) { v8f z = {}; ctx[t] = z; }

  for (int kc = 0; kc < q0 + 16; kc += 32) {  // causal: chunks of 32 keys
    // ---- scores: S = Q @ K^T  (two 16x16 n-tiles) ----
    v8f sc[2];
#pragma unroll
    for (int t = 0; t < 2; ++t) { v8f z = {}; sc[t] = z; }
#pragma unroll
    for (int kd = 0; kd < 2; ++kd) {
#pragma unroll
      for (int t = 0; t < 2; ++t) {
        Frag bk;
        const bf16* kp = Kp + (size_t)(kc + 16 * t + l16) * HDIM + 32 * kd + 8 * g;
        bk.h[0] = *(const v8bf*)kp;
        bk.h[1] = *(const v8bf*)(kp + 16);
        sc[t] = wmma_bf16(qa[kd].v, bk.v, sc[t]);
      }
    }
    // ---- causal mask (boundary chunk only) ----
    if (kc + 31 > q0) {
#pragma unroll
      for (int t = 0; t < 2; ++t) {
        const int key = kc + 16 * t + l16;
#pragma unroll
        for (int r = 0; r < 8; ++r) {
          const int q = q0 + r + 8 * g;
          if (key > q) sc[t][r] = -1e30f;
        }
      }
    }
    // ---- online softmax (rows live per 16-lane half; xor masks < 16) ----
#pragma unroll
    for (int r = 0; r < 8; ++r) {
      float cm = fmaxf(sc[0][r], sc[1][r]);
      cm = fmaxf(cm, __shfl_xor(cm, 1, 32));
      cm = fmaxf(cm, __shfl_xor(cm, 2, 32));
      cm = fmaxf(cm, __shfl_xor(cm, 4, 32));
      cm = fmaxf(cm, __shfl_xor(cm, 8, 32));
      const float nm   = fmaxf(mrow[r], cm);
      const float corr = __expf(mrow[r] - nm);
      const float p0   = __expf(sc[0][r] - nm);
      const float p1   = __expf(sc[1][r] - nm);
      float rs = p0 + p1;
      rs += __shfl_xor(rs, 1, 32);
      rs += __shfl_xor(rs, 2, 32);
      rs += __shfl_xor(rs, 4, 32);
      rs += __shfl_xor(rs, 8, 32);
      lrow[r] = lrow[r] * corr + rs;
      mrow[r] = nm;
#pragma unroll
      for (int t = 0; t < 4; ++t) ctx[t][r] *= corr;
      // D-layout -> LDS row-major (relayout for A fragment)
      sP[w][r + 8 * g][l16]      = (bf16)p0;
      sP[w][r + 8 * g][16 + l16] = (bf16)p1;
    }
    // ---- read P back as A fragment (DS ops are wave-in-order) ----
    Frag pa;
    {
      const __bf16* pp = &sP[w][l16][8 * g];
      pa.h[0] = *(const v8bf*)pp;
      pa.h[1] = *(const v8bf*)(pp + 16);
    }
    // ---- context += P(16x32) @ V(32x64) ----
#pragma unroll
    for (int t = 0; t < 4; ++t) {
      Frag bv;
      const bf16* vp = Vp + (size_t)(16 * t + l16) * SS + kc + 8 * g;
      bv.h[0] = *(const v8bf*)vp;
      bv.h[1] = *(const v8bf*)(vp + 16);
      ctx[t] = wmma_bf16(pa.v, bv.v, ctx[t]);
    }
  }

  // ---- normalize and store context as (B, S, H*HD) bf16 ----
#pragma unroll
  for (int r = 0; r < 8; ++r) {
    const float inv = 1.0f / lrow[r];
    const int q = q0 + r + 8 * g;
    const size_t base = (((size_t)b * SS + q) * HH + h) * HDIM;
#pragma unroll
    for (int t = 0; t < 4; ++t)
      Ctx[base + 16 * t + l16] = (bf16)(ctx[t][r] * inv);
  }
}

// ---------------------------------------------------------------------------
// Output projection: Ctx(4096x1024) @ WtO^T + b_o -> fp32 d_out
// ---------------------------------------------------------------------------
__global__ void __launch_bounds__(128)
out_gemm_kernel(const bf16* __restrict__ Cb, const bf16* __restrict__ Wt,
                const float* __restrict__ bo, float* __restrict__ out) {
  const int wave = blockIdx.x * (blockDim.x >> 5) + (threadIdx.x >> 5);
  const int NT = DD / 64;    // 16
  const int MT = MROWS / 64; // 64
  if (wave >= MT * NT) return;
  const int tj = wave % NT;
  const int ti = wave / NT;
  const int m0 = ti * 64, n0 = tj * 64;
  const int lane = threadIdx.x & 31;
  const int l16  = lane & 15;
  const int g    = lane >> 4;

  v8f acc[4][4];
#pragma unroll
  for (int i = 0; i < 4; ++i)
#pragma unroll
    for (int j = 0; j < 4; ++j) { v8f z = {}; acc[i][j] = z; }

  mma_64x64(Cb, Wt, m0, n0, DD, acc);

#pragma unroll
  for (int j = 0; j < 4; ++j) {
    const int n    = n0 + 16 * j + l16;
    const float bn = bo[n];
#pragma unroll
    for (int i = 0; i < 4; ++i) {
#pragma unroll
      for (int r = 0; r < 8; ++r) {
        const int m = m0 + 16 * i + r + 8 * g;
        out[(size_t)m * DD + n] = acc[i][j][r] + bn;
      }
    }
  }
}

// ---------------------------------------------------------------------------
// Host-side launcher
// ---------------------------------------------------------------------------
extern "C" void kernel_launch(void* const* d_in, const int* in_sizes, int n_in,
                              void* d_out, int out_size, void* d_ws, size_t ws_size,
                              hipStream_t stream) {
  const float* x    = (const float*)d_in[0];  // (B,S,D)
  const float* Wqkv = (const float*)d_in[1];  // (D, 3D)
  const float* bqkv = (const float*)d_in[2];  // (3D,)
  const float* Wo   = (const float*)d_in[3];  // (D, D)
  const float* bo   = (const float*)d_in[4];  // (D,)
  // d_in[5] = mask: causal tril, applied analytically in attn_kernel
  float* out = (float*)d_out;

  // Workspace carve-up (all chunks multiples of 256 B)
  char* ws = (char*)d_ws;
  size_t off = 0;
  bf16* Xbf  = (bf16*)(ws + off); off += (size_t)MROWS * DD * sizeof(bf16);       // 8 MB
  bf16* WtQ  = (bf16*)(ws + off); off += (size_t)NQKV * DD * sizeof(bf16);        // 6 MB
  bf16* WtO  = (bf16*)(ws + off); off += (size_t)DD * DD * sizeof(bf16);          // 2 MB
  bf16* Qb   = (bf16*)(ws + off); off += (size_t)BB * HH * SS * HDIM * sizeof(bf16); // 8 MB
  bf16* Kb   = (bf16*)(ws + off); off += (size_t)BB * HH * SS * HDIM * sizeof(bf16); // 8 MB
  bf16* Vt   = (bf16*)(ws + off); off += (size_t)BB * HH * HDIM * SS * sizeof(bf16); // 8 MB
  bf16* Ctx  = (bf16*)(ws + off); off += (size_t)MROWS * DD * sizeof(bf16);       // 8 MB
  (void)ws_size; (void)in_sizes; (void)n_in; (void)out_size;

  // 1) x -> bf16
  {
    int n = MROWS * DD;
    convert_bf16_kernel<<<(n + 255) / 256, 256, 0, stream>>>(x, Xbf, n);
  }
  // 2) W_qkv -> (3D, D) bf16 transposed ; W_o -> (D, D) bf16 transposed
  {
    int n = DD * NQKV;
    transpose_bf16_kernel<<<(n + 255) / 256, 256, 0, stream>>>(Wqkv, WtQ, DD, NQKV);
    n = DD * DD;
    transpose_bf16_kernel<<<(n + 255) / 256, 256, 0, stream>>>(Wo, WtO, DD, DD);
  }
  // 3) QKV projection GEMM (64x64 per wave, 4 waves/block)
  {
    const int waves = (MROWS / 64) * (NQKV / 64);  // 3072
    qkv_gemm_kernel<<<waves / 4, 128, 0, stream>>>(Xbf, WtQ, bqkv, Qb, Kb, Vt);
  }
  // 4) Causal flash attention (one wave per 16-query tile)
  {
    const int waves = BB * HH * (SS / 16);  // 4096
    attn_kernel<<<waves / 4, 128, 0, stream>>>(Qb, Kb, Vt, Ctx);
  }
  // 5) Output projection GEMM
  {
    const int waves = (MROWS / 64) * (DD / 64);  // 1024
    out_gemm_kernel<<<waves / 4, 128, 0, stream>>>(Ctx, WtO, bo, out);
  }
}
